// FedIGL_Node_GIN_63634235457723
// MI455X (gfx1250) — matrix operand order repaired and checked by reading, not compiled, not run on hardware
//
#include <hip/hip_runtime.h>
#include <hip/hip_bf16.h>

#define N_NODES 40000
#define N_EDGES 640000
#define NFEAT   500
#define NHID    128
#define NCLASS  10
#define K_TOP   320000
#define L_OFF_W (NHID * NHID)   // reference loop: only layer l = NLAYER-1 = 1 survives
#define L_OFF_B (NHID)
#define KP_MAX  512             // max padded K (NFEAT=500 -> 512)

typedef __attribute__((ext_vector_type(16))) __bf16 v16bf;
typedef __attribute__((ext_vector_type(8)))  float  v8f;

#if __has_builtin(__builtin_amdgcn_global_load_async_to_lds_b128) && \
    __has_builtin(__builtin_amdgcn_s_wait_asynccnt)
#define FG_ASYNC 1
typedef int fg_v4i __attribute__((vector_size(16)));
typedef __attribute__((address_space(1))) fg_v4i fg_gv4i;   // global int4
typedef __attribute__((address_space(3))) fg_v4i fg_lv4i;   // LDS int4
#endif

static __host__ int cdiv(int a, int b) { return (a + b - 1) / b; }

// ---------------------------------------------------------------------------
// Pack W[K x N] (f32, row-major) into bf16 WMMA B-fragment order:
//   Wp[ ((chunk*nct + ct)*32 + lane)*16 + e ] = W[(chunk*32 + kb(lane) + e + (e&8))*N + ct*16 + (lane&15)]
// so each lane's 16 elements are 32 contiguous bytes (two b128 loads).
// ---------------------------------------------------------------------------
__global__ void fg_pack_w(const float* __restrict__ W, __bf16* __restrict__ Wp,
                          int K, int N, int nct, int total)
{
    int idx = blockIdx.x * blockDim.x + threadIdx.x;
    if (idx >= total) return;
    int e    = idx & 15;
    int lane = (idx >> 4) & 31;
    int t    = idx >> 9;
    int ct   = t % nct;
    int chunk = t / nct;
    int col  = ct * 16 + (lane & 15);
    int gk   = chunk * 32 + ((lane < 16) ? 0 : 8) + ((e < 8) ? e : 8 + e);
    float v  = (gk < K && col < N) ? W[(size_t)gk * N + col] : 0.0f;
    Wp[idx] = (__bf16)v;
}

// ---------------------------------------------------------------------------
// bf16 WMMA GEMM:  C[M x N] = act( (A (+A2)) [M x K] @ W + bias )
// block = 256 threads = 8 waves; block tile = 16 rows; wave w -> cols 16w..16w+15.
// A strip staged once in LDS (f32, async-to-LDS when possible), W pre-packed bf16.
// ---------------------------------------------------------------------------
__global__ __launch_bounds__(256)
void fg_gemm(const float* __restrict__ A, const float* __restrict__ A2,
             const __bf16* __restrict__ Wp, const float* __restrict__ bias,
             float* __restrict__ C, int M, int K, int N, int nct, int relu)
{
    __shared__ float sAf[16 * KP_MAX];
    const int tid  = threadIdx.x;
    const int lane = tid & 31;
    const int wave = tid >> 5;
    const int row0 = blockIdx.x * 16;
    const int Kp   = (K + 31) & ~31;

    // ---- stage A strip (16 x Kp f32) into LDS, zero-padded ----
    {
        const int r  = tid >> 4;            // 16 threads per row
        const int gr = row0 + r;
#ifdef FG_ASYNC
        if (!A2 && gr < M) {
            float* Anc = const_cast<float*>(A);
            for (int k = (tid & 15) * 4; k < Kp; k += 64) {
                if (k + 4 <= K) {
                    __builtin_amdgcn_global_load_async_to_lds_b128(
                        (fg_gv4i*)(Anc + (size_t)gr * K + k),
                        (fg_lv4i*)(sAf + r * KP_MAX + k), 0, 0);
                } else {
                    float4 z = {0.f, 0.f, 0.f, 0.f};
                    *(float4*)(sAf + r * KP_MAX + k) = z;  // K is a multiple of 4 here
                }
            }
        } else
#endif
        {
            for (int k = (tid & 15) * 4; k < Kp; k += 64) {
                float4 v = {0.f, 0.f, 0.f, 0.f};
                if (gr < M) {
                    const float* ap = A + (size_t)gr * K;
                    const float* a2p = A2 ? (A2 + (size_t)gr * K) : nullptr;
                    float* vv = (float*)&v;
#pragma unroll
                    for (int j = 0; j < 4; ++j)
                        if (k + j < K) {
                            float t0 = ap[k + j];
                            if (a2p) t0 += a2p[k + j];
                            vv[j] = t0;
                        }
                }
                *(float4*)(sAf + r * KP_MAX + k) = v;
            }
        }
    }
#ifdef FG_ASYNC
    __builtin_amdgcn_s_wait_asynccnt(0);
#endif
    __syncthreads();

    // ---- WMMA main loop ----
    if (wave < nct) {   // wave-uniform: EXEC all ones inside
        const int arow = lane & 15;
        const int kbA  = (lane < 16) ? 0 : 8;
        const int bcol = wave * 16 + (lane & 15);
        const __bf16* wfrag = Wp + ((size_t)wave * 32 + lane) * 16;
        const size_t  wstep = (size_t)nct * 512;

        v8f acc = {};
        for (int kc = 0; kc < Kp; kc += 32) {
            const float* ap = sAf + arow * KP_MAX + kc + kbA;
            float af[16];
#pragma unroll
            for (int e = 0; e < 8; ++e) af[e] = ap[e];
#pragma unroll
            for (int e = 0; e < 8; ++e) af[8 + e] = ap[16 + e];
            v16bf a, b;
#pragma unroll
            for (int e = 0; e < 16; ++e) a[e] = (__bf16)af[e];
            b = *(const v16bf*)wfrag;
            wfrag += wstep;
            acc = __builtin_amdgcn_wmma_f32_16x16x32_bf16(
                false, a, false, b, (short)0, acc, false, false);
        }

        float bv = (bias && bcol < N) ? bias[bcol] : 0.0f;
#pragma unroll
        for (int r = 0; r < 8; ++r) {
            int grow = row0 + r + ((lane < 16) ? 0 : 8);
            if (grow < M && bcol < N) {
                float v = acc[r] + bv;
                if (relu) v = fmaxf(v, 0.0f);
                C[(size_t)grow * N + bcol] = v;
            }
        }
    }
}

// ---------------------------------------------------------------------------
// elementwise / graph kernels
// ---------------------------------------------------------------------------
__global__ void fg_scale(const float* __restrict__ in, float* __restrict__ out,
                         float s, int n)
{
    int i = blockIdx.x * blockDim.x + threadIdx.x;
    if (i < n) out[i] = s * in[i];
}

__global__ void fg_zero(unsigned* __restrict__ p, int n)
{
    int i = blockIdx.x * blockDim.x + threadIdx.x;
    if (i < n) p[i] = 0u;
}

// one wave per edge; lane handles 4 channels (32 lanes x 4 = 128)
__global__ __launch_bounds__(256)
void fg_scatter(const float* __restrict__ x0, const int* __restrict__ src,
                const int* __restrict__ dst, const float* __restrict__ w,
                float* __restrict__ agg, int E)
{
    int e = blockIdx.x * 8 + (threadIdx.x >> 5);
    if (e >= E) return;
    float wt = w ? w[e] : 1.0f;
    if (wt == 0.0f) return;           // pruned edges contribute nothing
    int lane = threadIdx.x & 31;
    int s = src[e], d = dst[e];
    float4 xv = ((const float4*)(x0 + (size_t)s * NHID))[lane];
    float* ap = agg + (size_t)d * NHID + lane * 4;
    atomicAdd(ap + 0, wt * xv.x);
    atomicAdd(ap + 1, wt * xv.y);
    atomicAdd(ap + 2, wt * xv.z);
    atomicAdd(ap + 3, wt * xv.w);
}

__global__ __launch_bounds__(256)
void fg_edge_dot(const float* __restrict__ inv, const int* __restrict__ src,
                 const int* __restrict__ dst, float* __restrict__ ew, int E)
{
    int e = blockIdx.x * 8 + (threadIdx.x >> 5);
    if (e >= E) return;
    int lane = threadIdx.x & 31;
    float4 a = ((const float4*)(inv + (size_t)src[e] * NHID))[lane];
    float4 b = ((const float4*)(inv + (size_t)dst[e] * NHID))[lane];
    float d = a.x * b.x + a.y * b.y + a.z * b.z + a.w * b.w;
#pragma unroll
    for (int off = 16; off; off >>= 1) d += __shfl_down(d, off, 32);
    if (lane == 0) ew[e] = d;
}

// order-preserving float<->uint for atomic max over possibly-negative floats
__device__ inline unsigned enc_f(float f)
{
    unsigned u = __float_as_uint(f);
    return (u >> 31) ? ~u : (u | 0x80000000u);
}
__device__ inline float dec_f(unsigned e)
{
    return (e >> 31) ? __uint_as_float(e & 0x7FFFFFFFu) : __uint_as_float(~e);
}

__global__ void fg_seg_max(const float* __restrict__ ew, const int* __restrict__ src,
                           unsigned* __restrict__ mxenc, int E)
{
    int e = blockIdx.x * blockDim.x + threadIdx.x;
    if (e < E) atomicMax(&mxenc[src[e]], enc_f(ew[e]));
}

__global__ void fg_exp_den(float* __restrict__ ew, const int* __restrict__ src,
                           const unsigned* __restrict__ mxenc,
                           float* __restrict__ den, int E)
{
    int e = blockIdx.x * blockDim.x + threadIdx.x;
    if (e >= E) return;
    int s = src[e];
    float ex = __expf(ew[e] - dec_f(mxenc[s]));
    ew[e] = ex;
    atomicAdd(&den[s], ex);
}

__global__ void fg_norm(float* __restrict__ ew, const int* __restrict__ src,
                        const float* __restrict__ den, int E)
{
    int e = blockIdx.x * blockDim.x + threadIdx.x;
    if (e < E) ew[e] = ew[e] / den[src[e]];
}

// ---------------------------------------------------------------------------
// radix-select top-k threshold (keys = positive float bits, MSB-first bytes)
// state: [0]=prefix [1]=kRem [2]=tie counter ; hist at +8 (256 bins)
// ---------------------------------------------------------------------------
#define HIST_OFF 8

__global__ void fg_topk_init(unsigned* __restrict__ st, unsigned ktop)
{
    int tid = threadIdx.x;
    if (tid == 0) { st[0] = 0u; st[1] = ktop; st[2] = 0u; }
    st[HIST_OFF + tid] = 0u;
}

__global__ void fg_hist(const float* __restrict__ soft, int E,
                        unsigned* __restrict__ st, int pass)
{
    int e = blockIdx.x * blockDim.x + threadIdx.x;
    if (e >= E) return;
    unsigned key  = __float_as_uint(soft[e]);   // soft >= 0: bits order as uint
    unsigned mask = (pass == 3) ? 0u : (0xFFFFFFFFu << ((pass + 1) * 8));
    if ((key & mask) == (st[0] & mask))
        atomicAdd(&st[HIST_OFF + ((key >> (pass * 8)) & 255u)], 1u);
}

__global__ void fg_select(unsigned* __restrict__ st, int pass)
{
    __shared__ unsigned h[256];
    int tid = threadIdx.x;
    h[tid] = st[HIST_OFF + tid];
    __syncthreads();
    if (tid == 0) {
        unsigned kRem = st[1];
        unsigned run = 0; int digit = 0; unsigned gtcount = 0;
        for (int d = 255; d >= 0; --d) {
            unsigned c = h[d];
            if (run + c >= kRem) { digit = d; gtcount = run; break; }
            run += c;
        }
        st[0] |= (unsigned)digit << (pass * 8);
        st[1] = kRem - gtcount;
    }
    __syncthreads();
    st[HIST_OFF + tid] = 0u;   // ready for next pass
}

__global__ void fg_mask(const float* __restrict__ soft, unsigned* __restrict__ st,
                        float* __restrict__ topw, float* __restrict__ remw, int E)
{
    int e = blockIdx.x * blockDim.x + threadIdx.x;
    if (e >= E) return;
    unsigned T = st[0], kRem = st[1];
    unsigned key = __float_as_uint(soft[e]);
    bool top;
    if (key > T)       top = true;
    else if (key == T) top = (atomicAdd(&st[2], 1u) < kRem);
    else               top = false;
    float v = soft[e];
    topw[e] = top ? v : 0.0f;
    remw[e] = top ? 0.0f : v;
}

// ---------------------------------------------------------------------------
extern "C" void kernel_launch(void* const* d_in, const int* in_sizes, int n_in,
                              void* d_out, int out_size, void* d_ws, size_t ws_size,
                              hipStream_t stream)
{
    const float* x      = (const float*)d_in[0];
    const int*   ei     = (const int*)d_in[1];
    const float* pre_W  = (const float*)d_in[2];
    const float* pre_b  = (const float*)d_in[3];
    const float* g1_W1  = (const float*)d_in[4];
    const float* g1_b1  = (const float*)d_in[5];
    const float* g1_W2  = (const float*)d_in[6];
    const float* g1_b2  = (const float*)d_in[7];
    const float* g3_W1  = (const float*)d_in[8];
    const float* g3_b1  = (const float*)d_in[9];
    const float* g3_W2  = (const float*)d_in[10];
    const float* g3_b2  = (const float*)d_in[11];
    const float* g4_W1  = (const float*)d_in[12];
    const float* g4_b1  = (const float*)d_in[13];
    const float* g4_W2  = (const float*)d_in[14];
    const float* g4_b2  = (const float*)d_in[15];
    const float* post_W = (const float*)d_in[16];
    const float* post_b = (const float*)d_in[17];
    const float* read_W = (const float*)d_in[18];
    const float* read_b = (const float*)d_in[19];

    const int* src = ei;
    const int* dst = ei + N_EDGES;

    // workspace carve-out (whole per-node working set is L2-resident)
    float* ws = (float*)d_ws;
    size_t off = 0;
    auto carve = [&](size_t n) { float* p = ws + off; off += (n + 63) & ~(size_t)63; return p; };
    float* x0   = carve((size_t)N_NODES * NHID);
    float* invb = carve((size_t)N_NODES * NHID);   // inv, later reused as x2
    float* agg  = carve((size_t)N_NODES * NHID);
    float* h1   = carve((size_t)N_NODES * NHID);
    float* x1   = carve((size_t)N_NODES * NHID);
    float* ewb  = carve(N_EDGES);                  // logits -> exp -> soft, in place
    float* topw = carve(N_EDGES);
    float* remw = carve(N_EDGES);
    unsigned* mxenc = (unsigned*)carve(N_NODES);
    float*    den   = carve(N_NODES);
    unsigned* st    = (unsigned*)carve(512);
    __bf16*   wp    = (__bf16*)carve(32768);       // packed weights, 64K bf16 max

    const int edge_waves = cdiv(N_EDGES, 8);
    const int edge_thr   = cdiv(N_EDGES, 256);
    const int feat_thr   = cdiv(N_NODES * NHID, 256);

    // pack-W + GEMM pair
    auto gemm = [&](const float* A, const float* A2, const float* W, const float* bias,
                    float* C, int M, int K, int N, int relu) {
        int nct = cdiv(N, 16);
        int total = cdiv(K, 32) * nct * 512;
        fg_pack_w<<<cdiv(total, 256), 256, 0, stream>>>(W, wp, K, N, nct, total);
        fg_gemm<<<cdiv(M, 16), 256, 0, stream>>>(A, A2, wp, bias, C, M, K, N, nct, relu);
    };

    // 1) x0 = x @ pre_W + pre_b
    gemm(x, nullptr, pre_W, pre_b, x0, N_NODES, NFEAT, NHID, 0);

    // GIN stack (only layer l=1 survives the reference loop):
    auto stack = [&](const float* W1, const float* b1, const float* W2, const float* b2,
                     const float* wts, float* outbuf) {
        fg_scale<<<feat_thr, 256, 0, stream>>>(x0, agg, 2.0f, N_NODES * NHID); // (2+eps)*x0
        fg_scatter<<<edge_waves, 256, 0, stream>>>(x0, src, dst, wts, agg, N_EDGES);
        gemm(agg, nullptr, W1 + L_OFF_W, b1 + L_OFF_B, h1, N_NODES, NHID, NHID, 1);
        gemm(h1, nullptr, W2 + L_OFF_W, b2 + L_OFF_B, outbuf, N_NODES, NHID, NHID, 1);
    };

    // 2) inv stack (g3, unit weights)
    stack(g3_W1, g3_b1, g3_W2, g3_b2, nullptr, invb);

    // 3) per-edge logits
    fg_edge_dot<<<edge_waves, 256, 0, stream>>>(invb, src, dst, ewb, N_EDGES);

    // 4) segment softmax by src
    fg_zero<<<cdiv(N_NODES, 256), 256, 0, stream>>>(mxenc, N_NODES);
    fg_zero<<<cdiv(N_NODES, 256), 256, 0, stream>>>((unsigned*)den, N_NODES);
    fg_seg_max<<<edge_thr, 256, 0, stream>>>(ewb, src, mxenc, N_EDGES);
    fg_exp_den<<<edge_thr, 256, 0, stream>>>(ewb, src, mxenc, den, N_EDGES);
    fg_norm<<<edge_thr, 256, 0, stream>>>(ewb, src, den, N_EDGES);

    // 5) top-k split via 4-pass radix select on float bits
    fg_topk_init<<<1, 256, 0, stream>>>(st, K_TOP);
    for (int p = 3; p >= 0; --p) {
        fg_hist<<<edge_thr, 256, 0, stream>>>(ewb, N_EDGES, st, p);
        fg_select<<<1, 256, 0, stream>>>(st, p);
    }
    fg_mask<<<edge_thr, 256, 0, stream>>>(ewb, st, topw, remw, N_EDGES);

    // 6) x1 (g1, top weights) and x2 (g4, remaining weights; reuse invb)
    stack(g1_W1, g1_b1, g1_W2, g1_b2, topw, x1);
    stack(g4_W1, g4_b1, g4_W2, g4_b2, remw, invb);

    // 7) h = relu((x1+x2) @ post_W + post_b)  (A2-fused add; write into agg)
    gemm(x1, invb, post_W, post_b, agg, N_NODES, NHID, NHID, 1);
    // 8) out = h @ read_W + read_b   (N=10, ld=10)
    gemm(agg, nullptr, read_W, read_b, (float*)d_out, N_NODES, NHID, NCLASS, 0);
}
